// Attention_16578573762580
// MI455X (gfx1250) — matrix operand order repaired and checked by reading, not compiled, not run on hardware
//
#include <hip/hip_runtime.h>

typedef __attribute__((ext_vector_type(16))) _Float16 v16h;
typedef __attribute__((ext_vector_type(8)))  _Float16 v8h;
typedef __attribute__((ext_vector_type(8)))  float    v8f;
typedef __attribute__((ext_vector_type(4)))  unsigned int v4u;
typedef __attribute__((ext_vector_type(8)))  int v8i;
typedef __attribute__((ext_vector_type(4)))  int v4i;

constexpr int kB = 8, kN = 1024, kD = 256, kH = 8, kHD = 32;
constexpr int kNQT = kN / 16;   // 64 k/q tiles of 16
constexpr int kNKC = kN / 32;   // 32 k chunks of 32
constexpr float kScale = 0.17677669529663687f;              // 1/sqrt(32)
constexpr float kNegMaskedScaled = -1.7677669529663687e19f; // -1e20 * scale

__device__ __forceinline__ v8f wmma16(v16h a, v16h b, v8f c) {
  // D = A(16x32 f16) * B(32x16 f16) + C(16x16 f32)
  return __builtin_amdgcn_wmma_f32_16x16x32_f16(false, a, false, b, (short)0, c,
                                                false, false);
}

// ---------------------------------------------------------------- converts
__global__ void cvt_f32_f16(const float* __restrict__ s, _Float16* __restrict__ d, int n) {
  int i = blockIdx.x * blockDim.x + threadIdx.x;
  if (i < n) d[i] = (_Float16)s[i];
}

// mask int32 (0/1) -> u8: quarters the dominant memory stream (33.5MB -> 8.4MB)
__global__ void cvt_mask_u8(const int* __restrict__ m, unsigned char* __restrict__ m8, int n4) {
  int i = blockIdx.x * blockDim.x + threadIdx.x;
  if (i < n4) {
    int4 v = ((const int4*)m)[i];
    uchar4 o;
    o.x = (unsigned char)v.x; o.y = (unsigned char)v.y;
    o.z = (unsigned char)v.z; o.w = (unsigned char)v.w;
    ((uchar4*)m8)[i] = o;
  }
}

// ------------------------------------------------------- QKV projection
// One wave per (b, h, 16-row tile, proj). tile = x_tile(16x32) @ W^T with two
// WMMAs, result written pre-swizzled into the exact per-lane fragment vectors
// the attention kernel consumes (32B contiguous per lane).
__global__ __launch_bounds__(32)
void qkv_project(const _Float16* __restrict__ x16,
                 const _Float16* __restrict__ wq16,
                 const _Float16* __restrict__ wk16,
                 const _Float16* __restrict__ wv16,
                 _Float16* __restrict__ qswz,
                 _Float16* __restrict__ kswz,
                 _Float16* __restrict__ vswz) {
  __shared__ __align__(32) _Float16 tile[16][32];
  const int lane = threadIdx.x;
  int id = blockIdx.x;
  const int proj = id % 3; id /= 3;
  const int qt = id % kNQT; id /= kNQT;
  const int h = id % kH;
  const int b = id / kH;
  const _Float16* __restrict__ W = (proj == 0) ? wq16 : (proj == 1 ? wk16 : wv16);

  // A fragment: lane<16 row=lane K{0..15}; lane>=16 row=lane-16 K{16..31}
  const int arow = qt * 16 + (lane & 15);
  const int ad0 = (lane < 16) ? 0 : 16;
  const _Float16* ap = x16 + ((size_t)(b * kN + arow)) * kD + h * kHD;
  v8h a0 = *(const v8h*)(ap + ad0);
  v8h a1 = *(const v8h*)(ap + ad0 + 8);
  v16h a;
#pragma unroll
  for (int i = 0; i < 8; i++) { a[i] = a0[i]; a[8 + i] = a1[i]; }

  // B fragments: B[d][e] = W[e][d]; lane<16: d=0..15, lane>=16: d=16..31
  const int ecol = lane & 15;
  const int bc0 = (lane < 16) ? 0 : 16;
  v16h bf0, bf1;
  {
    const _Float16* p = W + (size_t)ecol * kHD + bc0;
    v8h x0 = *(const v8h*)p, x1 = *(const v8h*)(p + 8);
#pragma unroll
    for (int i = 0; i < 8; i++) { bf0[i] = x0[i]; bf0[8 + i] = x1[i]; }
    p = W + (size_t)(16 + ecol) * kHD + bc0;
    v8h y0 = *(const v8h*)p, y1 = *(const v8h*)(p + 8);
#pragma unroll
    for (int i = 0; i < 8; i++) { bf1[i] = y0[i]; bf1[8 + i] = y1[i]; }
  }
  v8f z = {};
  v8f d0 = wmma16(a, bf0, z);   // e = 0..15
  v8f d1 = wmma16(a, bf1, z);   // e = 16..31

  const int mrow = (lane < 16) ? 0 : 8;
  const int ncol = lane & 15;
#pragma unroll
  for (int i = 0; i < 8; i++) {
    tile[mrow + i][ncol]      = (_Float16)d0[i];
    tile[mrow + i][16 + ncol] = (_Float16)d1[i];
  }
  __syncthreads();

  const size_t tidx = ((size_t)((b * kH + h) * kNQT + qt) * 32 + lane) * 16;
  if (proj == 0) {           // Q: A-fragment layout
    const int r = lane & 15, dd = (lane < 16) ? 0 : 16;
    v8h o0 = *(const v8h*)&tile[r][dd];
    v8h o1 = *(const v8h*)&tile[r][dd + 8];
    v16h o;
#pragma unroll
    for (int i = 0; i < 8; i++) { o[i] = o0[i]; o[8 + i] = o1[i]; }
    *(v16h*)(qswz + tidx) = o;
  } else if (proj == 1) {    // K: scores-B layout
    const int kcol = lane & 15, ds0 = (lane < 16) ? 0 : 16;
    v8h o0 = *(const v8h*)&tile[kcol][ds0];
    v8h o1 = *(const v8h*)&tile[kcol][ds0 + 8];
    v16h o;
#pragma unroll
    for (int i = 0; i < 8; i++) { o[i] = o0[i]; o[8 + i] = o1[i]; }
    *(v16h*)(kswz + tidx) = o;
  } else {                   // V: ctx-B layout per 32-k chunk (half lanes/tile)
    const int chunk = qt >> 1, half = qt & 1;
    const int dblk = lane >> 4, lcol = lane & 15;
    const int dcol = dblk * 16 + lcol;
    v16h o;
#pragma unroll
    for (int j = 0; j < 16; j++) o[j] = tile[j][dcol];
    const int LB = half * 16 + lcol;
    const size_t vidx = ((size_t)(((b * kH + h) * kNKC + chunk) * 2 + dblk) * 32 + LB) * 16;
    *(v16h*)(vswz + vidx) = o;
  }
}

// ----------------------------------------------------- fused flash attention
// One workgroup per (b, 16-row q tile); 8 waves = 8 heads (the renorm couples
// heads elementwise). Mask slab (16 rows x 1024 u8 = 16KB) is DMA'd into LDS
// once per workgroup by the Tensor Data Mover, so neither pass touches global
// memory for the mask.
__global__ __launch_bounds__(256)
void attention(const _Float16* __restrict__ qswz,
               const _Float16* __restrict__ kswz,
               const _Float16* __restrict__ vswz,
               const unsigned char* __restrict__ mask8,
               _Float16* __restrict__ ctx16) {
  __shared__ float accS[2][16][16];
  __shared__ __align__(32) _Float16 wbuf[kH][16][32];
  __shared__ __align__(64) unsigned char maskLds[16 * kN];  // 16KB
  const int lane = threadIdx.x & 31;
  const int h = threadIdx.x >> 5;
  const int qt = blockIdx.x % kNQT;
  const int b = blockIdx.x / kNQT;
  const int q0 = qt * 16;

  for (int i = threadIdx.x; i < 2 * 16 * 16; i += 256) ((float*)accS)[i] = 0.f;

  const unsigned char* gsrc = mask8 + (size_t)b * kN * kN + (size_t)q0 * kN;
#if __has_builtin(__builtin_amdgcn_tensor_load_to_lds) && __has_builtin(__builtin_amdgcn_s_wait_tensorcnt)
  if (h == 0) {
    // 1-D TDM descriptor: 16384 x 1B elements, contiguous.
    const unsigned long long ga = (unsigned long long)(uintptr_t)gsrc;
    const unsigned int ldsa = (unsigned int)(uintptr_t)&maskLds[0];
    v4u g0;
    g0[0] = 1u;                                        // count=1, user mode
    g0[1] = ldsa;                                      // lds_addr
    g0[2] = (unsigned int)(ga & 0xFFFFFFFFu);          // global_addr[31:0]
    g0[3] = (unsigned int)((ga >> 32) & 0x1FFFFFFu)    // global_addr[56:32]
            | (2u << 30);                              // type = 2 (image)
    v8i g1;
    g1[0] = 0;                       // workgroup_mask=0, data_size=0 (1B)
    g1[1] = (int)(16384u << 16);     // tensor_dim0[15:0] -> bits[63:48]
    g1[2] = (int)(1u << 16);         // tensor_dim0 hi=0; tensor_dim1=1
    g1[3] = (int)(16384u << 16);     // tensor_dim1 hi=0; tile_dim0=16384
    g1[4] = 1;                       // tile_dim1=1 (tile_dim2=0)
    g1[5] = 16384;                   // tensor_dim0_stride (low 32)
    g1[6] = 0;
    g1[7] = 0;
    v4i gz = {0, 0, 0, 0};
#if defined(__clang_major__) && __clang_major__ >= 23
    v8i gz8 = {0, 0, 0, 0, 0, 0, 0, 0};
    __builtin_amdgcn_tensor_load_to_lds(g0, g1, gz, gz, gz8, 0);
#else
    __builtin_amdgcn_tensor_load_to_lds(g0, g1, gz, gz, 0);
#endif
    __builtin_amdgcn_s_wait_tensorcnt((short)0);
  }
#else
  for (int i = threadIdx.x; i < (16 * kN) / 16; i += 256)
    ((uint4*)maskLds)[i] = ((const uint4*)gsrc)[i];
#endif

  const v16h aQ =
      *(const v16h*)(qswz + ((size_t)((b * kH + h) * kNQT + qt) * 32 + lane) * 16);

  const int mrow = (lane < 16) ? 0 : 8;
  const int ncol = lane & 15;

  float m[8], l[8];
#pragma unroll
  for (int i = 0; i < 8; i++) { m[i] = -3.0e38f; l[i] = 0.f; }

  const _Float16* kbase = kswz + (size_t)(b * kH + h) * kNQT * 512;
  const v8f z8 = {};

  __syncthreads();  // mask slab in LDS, accS zeroed

  // ---- pass 1: per-lane online max/sum over all key tiles
  for (int kt = 0; kt < kNQT; ++kt) {
    __builtin_prefetch(kbase + (size_t)(((kt + 1) & (kNQT - 1)) * 32 + lane) * 16, 0, 3);
    v16h kb = *(const v16h*)(kbase + (size_t)(kt * 32 + lane) * 16);
    v8f s = wmma16(aQ, kb, z8);
#pragma unroll
    for (int i = 0; i < 8; i++) {
      const int mv = maskLds[(mrow + i) * kN + kt * 16 + ncol];
      const float sv = mv ? s[i] * kScale : kNegMaskedScaled;
      if (sv > m[i]) { l[i] = l[i] * __expf(m[i] - sv) + 1.0f; m[i] = sv; }
      else           { l[i] += __expf(sv - m[i]); }
    }
  }
  // merge stats across the 16 lanes holding each row (masks 1..8 stay in-half)
#pragma unroll
  for (int off = 1; off < 16; off <<= 1) {
#pragma unroll
    for (int i = 0; i < 8; i++) {
      const float mo = __shfl_xor(m[i], off, 32);
      const float lo = __shfl_xor(l[i], off, 32);
      const float M = fmaxf(m[i], mo);
      l[i] = l[i] * __expf(m[i] - M) + lo * __expf(mo - M);
      m[i] = M;
    }
  }

  v8f c0 = {}, c1 = {};
  const _Float16* vbase = vswz + (size_t)(b * kH + h) * kNKC * 1024;

  // ---- pass 2: recompute p, cross-head renorm via ds_add_f32, ctx WMMA
  for (int kt = 0; kt < kNQT; ++kt) {
    const int buf = kt & 1;
    __builtin_prefetch(vbase + (size_t)((kt >> 1) * 64 + lane) * 16, 0, 3);
    v16h kb = *(const v16h*)(kbase + (size_t)(kt * 32 + lane) * 16);
    v8f s = wmma16(aQ, kb, z8);
    float a8[8];
#pragma unroll
    for (int i = 0; i < 8; i++) {
      const int mv = maskLds[(mrow + i) * kN + kt * 16 + ncol];
      const float sv = mv ? s[i] * kScale : kNegMaskedScaled;
      a8[i] = __expf(sv - m[i]) / l[i];
      atomicAdd(&accS[buf][mrow + i][ncol], a8[i]);  // ds_add_f32, 8 heads
    }
    __syncthreads();
#pragma unroll
    for (int i = 0; i < 8; i++) {
      const float w = a8[i] / accS[buf][mrow + i][ncol] + 1e-9f;
      wbuf[h][mrow + i][buf * 16 + ncol] = (_Float16)w;
    }
    __syncthreads();
    if (h == 0) {  // recycle this buffer for kt+2 (safe: barrier at kt+1)
#pragma unroll
      for (int i = 0; i < 8; i++) ((float*)accS[buf])[lane + 32 * i] = 0.f;
    }
    if (buf) {  // every 32 keys: w(16x32) @ V(32x32) via two WMMAs
      const int chunk = kt >> 1;
      const int r = lane & 15, kk = (lane < 16) ? 0 : 16;
      v8h w0 = *(const v8h*)&wbuf[h][r][kk];
      v8h w1 = *(const v8h*)&wbuf[h][r][kk + 8];
      v16h wa;
#pragma unroll
      for (int i = 0; i < 8; i++) { wa[i] = w0[i]; wa[8 + i] = w1[i]; }
      v16h vb0 = *(const v16h*)(vbase + (size_t)((chunk * 2 + 0) * 32 + lane) * 16);
      v16h vb1 = *(const v16h*)(vbase + (size_t)((chunk * 2 + 1) * 32 + lane) * 16);
      c0 = wmma16(wa, vb0, c0);
      c1 = wmma16(wa, vb1, c1);
    }
  }
#pragma unroll
  for (int i = 0; i < 8; i++) {
    const size_t ro = (size_t)(b * kN + q0 + mrow + i) * kD + h * kHD;
    ctx16[ro + ncol]      = (_Float16)c0[i];
    ctx16[ro + 16 + ncol] = (_Float16)c1[i];
  }
}

// ------------------------------------------------------- output projection
// out(8192x256) = ctx16 @ Wo^T, one wave per 16x16 tile, K chunked by 32.
__global__ __launch_bounds__(32)
void out_project(const _Float16* __restrict__ ctx16,
                 const _Float16* __restrict__ wo16,
                 float* __restrict__ out) {
  const int lane = threadIdx.x;
  const int et = blockIdx.x & 15;
  const int mt = blockIdx.x >> 4;
  const int r = mt * 16 + (lane & 15);
  const int e = et * 16 + (lane & 15);
  const int ao = (lane < 16) ? 0 : 16;
  const int bo = (lane < 16) ? 0 : 16;
  v8f acc = {};
  for (int c = 0; c < 8; ++c) {
    const _Float16* ap = ctx16 + (size_t)r * kD + c * 32 + ao;
    v8h a0 = *(const v8h*)ap, a1 = *(const v8h*)(ap + 8);
    v16h a;
#pragma unroll
    for (int i = 0; i < 8; i++) { a[i] = a0[i]; a[8 + i] = a1[i]; }
    const _Float16* bp = wo16 + (size_t)e * kD + c * 32 + bo;
    v8h b0 = *(const v8h*)bp, b1 = *(const v8h*)(bp + 8);
    v16h bb;
#pragma unroll
    for (int i = 0; i < 8; i++) { bb[i] = b0[i]; bb[8 + i] = b1[i]; }
    acc = wmma16(a, bb, acc);
  }
  const int mrow = (lane < 16) ? 0 : 8, nc = lane & 15;
#pragma unroll
  for (int i = 0; i < 8; i++)
    out[(size_t)(mt * 16 + mrow + i) * kD + et * 16 + nc] = acc[i];
}

// ------------------------------------------------------------------ launch
extern "C" void kernel_launch(void* const* d_in, const int* in_sizes, int n_in,
                              void* d_out, int out_size, void* d_ws, size_t ws_size,
                              hipStream_t stream) {
  (void)in_sizes; (void)n_in; (void)out_size; (void)ws_size;
  const float* x  = (const float*)d_in[0];
  const int* mask = (const int*)d_in[1];
  const float* Wq = (const float*)d_in[2];
  const float* Wk = (const float*)d_in[3];
  const float* Wv = (const float*)d_in[4];
  const float* Wo = (const float*)d_in[5];
  float* out = (float*)d_out;

  char* ws = (char*)d_ws;
  size_t off = 0;
  auto grabB = [&](size_t bytes) -> char* {
    char* p = ws + off;
    off += (bytes + 255) & ~(size_t)255;
    return p;
  };
  _Float16* x16  = (_Float16*)grabB((size_t)kB * kN * kD * 2);
  _Float16* wq16 = (_Float16*)grabB(kHD * kHD * 2);
  _Float16* wk16 = (_Float16*)grabB(kHD * kHD * 2);
  _Float16* wv16 = (_Float16*)grabB(kHD * kHD * 2);
  _Float16* wo16 = (_Float16*)grabB((size_t)kD * kD * 2);
  _Float16* qswz = (_Float16*)grabB((size_t)kB * kH * kN * kHD * 2);
  _Float16* kswz = (_Float16*)grabB((size_t)kB * kH * kN * kHD * 2);
  _Float16* vswz = (_Float16*)grabB((size_t)kB * kH * kN * kHD * 2);
  _Float16* ctx16 = (_Float16*)grabB((size_t)kB * kN * kD * 2);
  unsigned char* mask8 = (unsigned char*)grabB((size_t)kB * kN * kN);

  const int nx = kB * kN * kD;
  cvt_f32_f16<<<(nx + 255) / 256, 256, 0, stream>>>(x, x16, nx);
  cvt_f32_f16<<<4, 256, 0, stream>>>(Wq, wq16, kHD * kHD);
  cvt_f32_f16<<<4, 256, 0, stream>>>(Wk, wk16, kHD * kHD);
  cvt_f32_f16<<<4, 256, 0, stream>>>(Wv, wv16, kHD * kHD);
  cvt_f32_f16<<<(kD * kD + 255) / 256, 256, 0, stream>>>(Wo, wo16, kD * kD);
  const int nm4 = kB * kN * kN / 4;
  cvt_mask_u8<<<(nm4 + 255) / 256, 256, 0, stream>>>(mask, mask8, nm4);

  qkv_project<<<kB * kH * kNQT * 3, 32, 0, stream>>>(x16, wq16, wk16, wv16,
                                                     qswz, kswz, vswz);
  attention<<<kB * kNQT, 256, 0, stream>>>(qswz, kswz, vswz, mask8, ctx16);
  out_project<<<(kB * kN / 16) * (kD / 16), 32, 0, stream>>>(ctx16, wo16, out);
}
